// ClusteringLayer_78829829750998
// MI455X (gfx1250) — matrix operand order repaired and checked by reading, not compiled, not run on hardware
//
#include <hip/hip_runtime.h>
#include <hip/hip_bf16.h>

typedef __attribute__((ext_vector_type(8)))  float          v8f;
typedef __attribute__((ext_vector_type(4)))  float          v4f;
typedef __attribute__((ext_vector_type(16))) __bf16         v16bf;
typedef __attribute__((ext_vector_type(8)))  unsigned short v8u16;
typedef __attribute__((ext_vector_type(16))) unsigned short v16u16;

#define KC  256   // clusters
#define DIM 256   // feature dim
#define MT  64    // output rows per workgroup

static __device__ __forceinline__ unsigned short bf16_rne(float f) {
    unsigned int u = __float_as_uint(f);
    unsigned int r = u + 0x7FFFu + ((u >> 16) & 1u);
    return (unsigned short)(r >> 16);
}
static __device__ __forceinline__ v16bf cat16(v8u16 a, v8u16 b) {
    v16u16 u = __builtin_shufflevector(a, b, 0,1,2,3,4,5,6,7,8,9,10,11,12,13,14,15);
    return __builtin_bit_cast(v16bf, u);
}

// ---------------------------------------------------------------------------
// Prep: round C (KC x DIM f32) to a single bf16 plane + per-row ||c||^2.
// (dist is dominated by the exact f32 norms)
// ---------------------------------------------------------------------------
__global__ __launch_bounds__(DIM) void prep_centers_kernel(
    const float* __restrict__ C,
    unsigned short* __restrict__ Cb,
    float* __restrict__ csq)
{
    __shared__ float red[DIM];
    const int row = blockIdx.x;
    const int t   = threadIdx.x;

    float x = C[(size_t)row * DIM + t];
    Cb[(size_t)row * DIM + t] = bf16_rne(x);

    red[t] = x * x;
    __syncthreads();
    #pragma unroll
    for (int s = DIM / 2; s > 0; s >>= 1) {
        if (t < s) red[t] += red[t + s];
        __syncthreads();
    }
    if (t == 0) csq[row] = red[0];
}

// ---------------------------------------------------------------------------
// Main: one workgroup (256 thr = 8 waves) computes MT=64 rows x 256 cols of q.
// Each wave: 2 column tiles x 4 row tiles, bf16x2 split (z = hi+lo) GEMM via
// v_wmma_f32_16x16x32_bf16; B fragments register-reused across 4 row tiles.
// ---------------------------------------------------------------------------
__global__ __launch_bounds__(256) void cluster_q_kernel(
    const float* __restrict__ z,
    const unsigned short* __restrict__ Cb,
    const float* __restrict__ csq_g,
    float* __restrict__ out)
{
    __shared__ unsigned short zh[MT][264];   // bf16 hi plane, padded stride
    __shared__ unsigned short zl[MT][264];   // bf16 lo plane
    __shared__ float csq[KC];
    __shared__ float part[MT][4];
    __shared__ float zsq[MT];
    __shared__ float wsum[MT][8];
    __shared__ float invs[MT];

    const int t    = threadIdx.x;
    const int wave = t >> 5;
    const int lane = t & 31;
    const int m0   = blockIdx.x * MT;

    // ---- stage z tile -> LDS (f32 -> bf16 hi/lo), packed b128 stores ----
    {
        const int r  = t >> 2;          // row 0..63
        const int c0 = (t & 3) * 64;    // 64-col segment
        const float* zp = z + (size_t)(m0 + r) * DIM + c0;
        float a = 0.f;
        #pragma unroll
        for (int i = 0; i < 8; ++i) {   // 8 groups of 8 floats
            v4f va = *(const v4f*)(zp + i * 8);
            v4f vb = *(const v4f*)(zp + i * 8 + 4);
            v8u16 hp, lp;
            #pragma unroll
            for (int j = 0; j < 8; ++j) {
                float x = (j < 4) ? va[j] : vb[j - 4];
                // truncation split: hi = top 16 bits (exact recon), lo = RNE(x - hi)
                unsigned int u  = __float_as_uint(x);
                unsigned int hu = u & 0xFFFF0000u;
                float lo = x - __uint_as_float(hu);
                hp[j] = (unsigned short)(hu >> 16);
                lp[j] = bf16_rne(lo);
                a += x * x;
            }
            *(v8u16*)&zh[r][c0 + i * 8] = hp;
            *(v8u16*)&zl[r][c0 + i * 8] = lp;
        }
        part[r][t & 3] = a;
        csq[t] = csq_g[t];
    }
    __syncthreads();
    if (t < MT) zsq[t] = part[t][0] + part[t][1] + part[t][2] + part[t][3];
    __syncthreads();

    // ---- WMMA main loop ----
    const int n    = lane & 15;     // A row within tile / B column / D N-index
    const int hs   = lane >> 4;     // half-wave select
    const int col0 = wave * 32;
    const int col1 = col0 + 16;

    const unsigned short* b0p = Cb + (size_t)(col0 + n) * DIM;
    const unsigned short* b1p = Cb + (size_t)(col1 + n) * DIM;

    v8f acc[8] = {};   // [rt*2 + ct]

    #pragma unroll
    for (int kk = 0; kk < 8; ++kk) {
        const int kb = kk * 32;
        // B (32x16 bf16): lanes 0-15 hold K=kb..kb+15, lanes 16-31 K=kb+16..kb+31
        const int ko = kb + hs * 16;
        v16bf B0 = __builtin_bit_cast(v16bf, *(const v16u16*)(b0p + ko));
        v16bf B1 = __builtin_bit_cast(v16bf, *(const v16u16*)(b1p + ko));

        #pragma unroll
        for (int rt = 0; rt < 4; ++rt) {
            const int r = rt * 16 + n;
            // A (16x32 bf16): lane<16 -> K kb..kb+15; lane>=16 -> K kb+16..kb+31
            v16bf Ahi = cat16(*(const v8u16*)&zh[r][kb + hs * 16],
                              *(const v8u16*)&zh[r][kb + 8 + hs * 16]);
            v16bf Alo = cat16(*(const v8u16*)&zl[r][kb + hs * 16],
                              *(const v8u16*)&zl[r][kb + 8 + hs * 16]);

            acc[rt*2+0] = __builtin_amdgcn_wmma_f32_16x16x32_bf16(false, Ahi, false, B0, (short)0, acc[rt*2+0], false, false);
            acc[rt*2+0] = __builtin_amdgcn_wmma_f32_16x16x32_bf16(false, Alo, false, B0, (short)0, acc[rt*2+0], false, false);
            acc[rt*2+1] = __builtin_amdgcn_wmma_f32_16x16x32_bf16(false, Ahi, false, B1, (short)0, acc[rt*2+1], false, false);
            acc[rt*2+1] = __builtin_amdgcn_wmma_f32_16x16x32_bf16(false, Alo, false, B1, (short)0, acc[rt*2+1], false, false);
        }
    }

    // ---- epilogue: dist -> q (in place over acc) -> row-normalize ----
    // C/D layout: vgpr v, lane: M = v + 8*(lane>=16), N = lane%16
    const float cs0 = csq[col0 + n];
    const float cs1 = csq[col1 + n];
    #pragma unroll
    for (int rt = 0; rt < 4; ++rt) {
        v4f zqa = *(const v4f*)&zsq[rt * 16 + hs * 8];
        v4f zqb = *(const v4f*)&zsq[rt * 16 + hs * 8 + 4];
        #pragma unroll
        for (int v = 0; v < 8; ++v) {
            const int m = rt * 16 + v + hs * 8;
            const float zq = (v < 4) ? zqa[v] : zqb[v - 4];
            float d0 = fmaxf(zq - 2.f * acc[rt*2+0][v] + cs0, 0.f);
            float d1 = fmaxf(zq - 2.f * acc[rt*2+1][v] + cs1, 0.f);
            float q0 = 1.f / (1.f + d0);   // ALPHA==1 -> exponent (ALPHA+1)/2 == 1
            float q1 = 1.f / (1.f + d1);
            acc[rt*2+0][v] = q0;
            acc[rt*2+1][v] = q1;
            float rs = q0 + q1;
            rs += __shfl_xor(rs, 1, 32);
            rs += __shfl_xor(rs, 2, 32);
            rs += __shfl_xor(rs, 4, 32);
            rs += __shfl_xor(rs, 8, 32);   // sum over N within each 16-lane half
            if (n == 0) wsum[m][wave] = rs;
        }
    }
    __syncthreads();
    if (t < MT) {
        float s = 0.f;
        #pragma unroll
        for (int w = 0; w < 8; ++w) s += wsum[t][w];
        invs[t] = 1.f / s;
    }
    __syncthreads();
    #pragma unroll
    for (int rt = 0; rt < 4; ++rt) {
        v4f iva = *(const v4f*)&invs[rt * 16 + hs * 8];
        v4f ivb = *(const v4f*)&invs[rt * 16 + hs * 8 + 4];
        #pragma unroll
        for (int v = 0; v < 8; ++v) {
            const int m = rt * 16 + v + hs * 8;
            const float iv = (v < 4) ? iva[v] : ivb[v - 4];
            float* o = out + (size_t)(m0 + m) * KC;
            o[col0 + n] = acc[rt*2+0][v] * iv;
            o[col1 + n] = acc[rt*2+1][v] * iv;
        }
    }
}

// ---------------------------------------------------------------------------
extern "C" void kernel_launch(void* const* d_in, const int* in_sizes, int n_in,
                              void* d_out, int out_size, void* d_ws, size_t ws_size,
                              hipStream_t stream)
{
    const float* z = (const float*)d_in[0];            // (N, 256) f32
    const float* C = (const float*)d_in[1];            // (256, 256) f32
    float* out = (float*)d_out;                        // (N, 256) f32
    const int N = in_sizes[0] / DIM;

    unsigned short* Cb = (unsigned short*)d_ws;                  // 128 KB
    float* csq = (float*)(Cb + (size_t)KC * DIM);                // 1 KB

    prep_centers_kernel<<<KC, DIM, 0, stream>>>(C, Cb, csq);
    cluster_q_kernel<<<N / MT, 256, 0, stream>>>(z, Cb, csq, out);
}
